// CausalSelfAttention_27127013441644
// MI455X (gfx1250) — matrix operand order repaired and checked by reading, not compiled, not run on hardware
//
#include <hip/hip_runtime.h>
#include <hip/hip_bf16.h>
#include <stdint.h>

// ---------------------------------------------------------------------------
// Problem constants (match reference)
// ---------------------------------------------------------------------------
#define B_    2
#define T_    2048
#define C_    1024
#define NH    16
#define KVH   4
#define HD    64
#define MROWS (B_ * T_)          // 4096

typedef __attribute__((ext_vector_type(16))) __bf16 v16bf;
typedef __attribute__((ext_vector_type(8)))  float  v8f;
typedef __attribute__((ext_vector_type(4)))  unsigned int u32x4;
typedef __attribute__((ext_vector_type(8)))  int   i32x8;
typedef __attribute__((ext_vector_type(4)))  int   i32x4;

#if __has_builtin(__builtin_amdgcn_tensor_load_to_lds) && __has_builtin(__builtin_amdgcn_s_wait_tensorcnt)
#define HAVE_TDM 1
#else
#define HAVE_TDM 0
#endif

struct U32B { uint4 a, b; };     // 32 bytes == one v16bf fragment

__device__ __forceinline__ v16bf mkfrag(uint4 a, uint4 b) {
  U32B t; t.a = a; t.b = b;
  return __builtin_bit_cast(v16bf, t);
}
__device__ __forceinline__ uint4 ld4(const unsigned short* p) {
  return *(const uint4*)p;       // caller guarantees 16B alignment
}
__device__ __forceinline__ unsigned short f2bf(float f) {
  unsigned u = __builtin_bit_cast(unsigned, f);
  u += 0x7fffu + ((u >> 16) & 1u);           // round-to-nearest-even
  return (unsigned short)(u >> 16);
}
__device__ __forceinline__ v8f wmma_bf16(v16bf a, v16bf b, v8f c) {
  // D = A(16x32) * B(32x16) + C(16x16 f32)
  return __builtin_amdgcn_wmma_f32_16x16x32_bf16(
      false, a, false, b, (short)0, c, false, false);
}

#if HAVE_TDM
// ---------------------------------------------------------------------------
// TDM 2D tile load: global (row-major, 2-byte elems) -> LDS, with pad of
// 4 DWORDs (16B) after every 16 DWORDs (64B) => 80B padded rows (LDP=40).
// Descriptor packing per CDNA5 ISA (D# groups 0/1).
// This toolchain's builtin takes 6 args (g0, g1, g2, g3-as-i32x4, extra i32x8
// group slot, cpol); unused groups are zero (tile_dim2/3 = 0 => 2D tile).
// ---------------------------------------------------------------------------
__device__ __forceinline__ void tdm_load_2d(
    unsigned lds_off, const void* gptr,
    unsigned tile_x, unsigned tile_y,
    unsigned long long row_stride_elems,
    unsigned long long tensor_x, unsigned long long tensor_y) {
  unsigned long long ga = (unsigned long long)gptr;
  u32x4 g0;
  g0[0] = 1u;                                  // count=1 (user descriptor)
  g0[1] = lds_off;                             // LDS byte address
  g0[2] = (unsigned)ga;                        // global_addr[31:0]
  g0[3] = (unsigned)(ga >> 32) | (2u << 30);   // global_addr[56:32] | type=2
  i32x8 g1;
  // data_size=1 (2B), pad_enable, pad_interval=3 (16 DW), pad_amount=3 (4 DW)
  g1[0] = (int)((1u << 16) | (1u << 20) | (3u << 22) | (3u << 25));
  g1[1] = (int)((unsigned)(tensor_x & 0xffffu) << 16);
  g1[2] = (int)(((unsigned)(tensor_x >> 16) & 0xffffu) |
                ((unsigned)(tensor_y & 0xffffu) << 16));
  g1[3] = (int)(((unsigned)(tensor_y >> 16) & 0xffffu) | (tile_x << 16));
  g1[4] = (int)(tile_y & 0xffffu);             // tile_dim1; tile_dim2=0
  g1[5] = (int)(unsigned)(row_stride_elems & 0xffffffffull);
  g1[6] = (int)(unsigned)((row_stride_elems >> 32) & 0xffffull);
  g1[7] = 0;
  i32x4 z4 = {0, 0, 0, 0};
  i32x8 z8 = {0, 0, 0, 0, 0, 0, 0, 0};
  __builtin_amdgcn_tensor_load_to_lds(g0, g1, z4, z4, z8, 0);
}
#endif

// ---------------------------------------------------------------------------
// 1) fp32 -> bf16 cast (row-major preserved)
// ---------------------------------------------------------------------------
__global__ __launch_bounds__(256) void cast_bf16_kernel(
    const float* __restrict__ in, unsigned short* __restrict__ out, long n) {
  long i = (long)blockIdx.x * 256 + threadIdx.x;
  if (i < n) out[i] = f2bf(in[i]);
}

// ---------------------------------------------------------------------------
// 2) W[K][N] fp32 -> Wt[N][K] bf16 (transpose+cast so GEMM B streams along K)
// ---------------------------------------------------------------------------
__global__ __launch_bounds__(256) void wtrans_kernel(
    const float* __restrict__ W, unsigned short* __restrict__ Wt, int K, int N) {
  long i = (long)blockIdx.x * 256 + threadIdx.x;   // output index n*K + k
  if (i >= (long)K * N) return;
  int k = (int)(i % K);
  int n = (int)(i / K);
  Wt[i] = f2bf(W[(size_t)k * N + n]);
}

// ---------------------------------------------------------------------------
// 3) GEMM: C[M][N] f32 = A[M][K] bf16 (row-major) * Bt[N][K] bf16
//    Block tile 128x128, 8 waves (4x2), 32x64 per wave, BK=32.
//    Tiles staged into double-buffered padded LDS by the Tensor Data Mover
//    (wave 0 issues descriptors; TENSORcnt + barrier synchronize).
//    M%128==0, N%128==0, K%32==0 (true for all uses here).
// ---------------------------------------------------------------------------
#define GBM 128
#define GBN 128
#define GBK 32
#define LDP 40                    // padded LDS row: 32 elems + 8 (80B, 16B-aligned)

__global__ __launch_bounds__(256) void gemm_bf16_tn_kernel(
    const unsigned short* __restrict__ A, const unsigned short* __restrict__ Bt,
    float* __restrict__ C, int M, int N, int K) {
  __shared__ __align__(16) unsigned short As[2][GBM * LDP];
  __shared__ __align__(16) unsigned short Bs[2][GBN * LDP];

  const int tid  = threadIdx.x;
  const int lane = tid & 31;
  const int wave = tid >> 5;         // 0..7
  const int wm   = wave >> 1;        // 0..3  (rows: wm*32)
  const int wn   = wave & 1;         // 0..1  (cols: wn*64)
  const int half = lane >> 4;
  const int l16  = lane & 15;

  const int m0 = blockIdx.y * GBM;
  const int n0 = blockIdx.x * GBN;
  const int nt = K / GBK;

  const unsigned short* Abase = A + (size_t)m0 * K;
  const unsigned short* Bbase = Bt + (size_t)n0 * K;

  v8f cc[2][4];
#pragma unroll
  for (int r = 0; r < 2; r++)
#pragma unroll
    for (int j = 0; j < 4; j++) cc[r][j] = (v8f){};

#if HAVE_TDM
  const unsigned ldsA[2] = {(unsigned)(uintptr_t)&As[0][0],
                            (unsigned)(uintptr_t)&As[1][0]};
  const unsigned ldsB[2] = {(unsigned)(uintptr_t)&Bs[0][0],
                            (unsigned)(uintptr_t)&Bs[1][0]};
  if (wave == 0) {
    tdm_load_2d(ldsA[0], Abase, GBK, GBM, (unsigned long long)K,
                (unsigned long long)K, (unsigned long long)GBM);
    tdm_load_2d(ldsB[0], Bbase, GBK, GBN, (unsigned long long)K,
                (unsigned long long)K, (unsigned long long)GBN);
    __builtin_amdgcn_s_wait_tensorcnt(0);
  }
  __syncthreads();
#endif

  for (int i = 0; i < nt; ++i) {
    const int cur = i & 1;
#if HAVE_TDM
    if (wave == 0 && (i + 1) < nt) {  // prefetch next tile into other buffer
      size_t go = (size_t)(i + 1) * GBK * 2;
      tdm_load_2d(ldsA[cur ^ 1], (const char*)Abase + go, GBK, GBM,
                  (unsigned long long)K, (unsigned long long)K,
                  (unsigned long long)GBM);
      tdm_load_2d(ldsB[cur ^ 1], (const char*)Bbase + go, GBK, GBN,
                  (unsigned long long)K, (unsigned long long)K,
                  (unsigned long long)GBN);
    }
#else
    { // manual staging fallback: 256 threads x 16 bf16 per tile
      int k0 = i * GBK;
      int row = tid >> 1, hb = tid & 1;
      const unsigned short* g = Abase + (size_t)row * K + k0 + hb * 16;
      unsigned short* d = &As[cur][row * LDP + hb * 16];
      uint4 u0 = ld4(g), u1 = ld4(g + 8);
      *(uint4*)d = u0; *(uint4*)(d + 8) = u1;
      const unsigned short* g2 = Bbase + (size_t)row * K + k0 + hb * 16;
      unsigned short* d2 = &Bs[cur][row * LDP + hb * 16];
      uint4 w0 = ld4(g2), w1 = ld4(g2 + 8);
      *(uint4*)d2 = w0; *(uint4*)(d2 + 8) = w1;
    }
    __syncthreads();
#endif

    const unsigned short* as = As[cur];
    const unsigned short* bs = Bs[cur];

    // A fragments (rows wm*32 .. +31)
    const unsigned short* ar0 = as + (wm * 32 + l16) * LDP;
    v16bf a0 = mkfrag(ld4(ar0 + half * 8), ld4(ar0 + 16 + half * 8));
    const unsigned short* ar1 = as + (wm * 32 + 16 + l16) * LDP;
    v16bf a1 = mkfrag(ld4(ar1 + half * 8), ld4(ar1 + 16 + half * 8));
    // B fragments (cols wn*64 .. +63, four 16-col tiles)
#pragma unroll
    for (int j = 0; j < 4; j++) {
      const unsigned short* br = bs + (wn * 64 + j * 16 + l16) * LDP + half * 8;
      v16bf bj = mkfrag(ld4(br), ld4(br + 16));
      cc[0][j] = wmma_bf16(a0, bj, cc[0][j]);
      cc[1][j] = wmma_bf16(a1, bj, cc[1][j]);
    }

#if HAVE_TDM
    if (wave == 0) __builtin_amdgcn_s_wait_tensorcnt(0);
#endif
    __syncthreads();
  }

#pragma unroll
  for (int r2 = 0; r2 < 2; r2++)
#pragma unroll
    for (int r = 0; r < 8; r++) {
      int mm = m0 + wm * 32 + r2 * 16 + r + 8 * half;
      float* cr = C + (size_t)mm * N + n0 + wn * 64;
#pragma unroll
      for (int j = 0; j < 4; j++) cr[j * 16 + l16] = cc[r2][j][r];
    }
}

// ---------------------------------------------------------------------------
// 4) RoPE + RMSNorm(no scale) + head transpose + bf16 cast.
//    in : [B,T,nheads,64] fp32   out: [B,nheads,T,64] bf16
//    One wave per (b,t,h) row; lane j owns rotation pair (j, j+32).
//    outscale folds attention scale * log2(e) into Q (1.0 for K).
// ---------------------------------------------------------------------------
__global__ __launch_bounds__(256) void rope_rms_kernel(
    const float* __restrict__ in, const float* __restrict__ cosb,
    const float* __restrict__ sinb, unsigned short* __restrict__ out,
    int nheads, float outscale) {
  const int lane = threadIdx.x & 31;
  const int w    = threadIdx.x >> 5;
  long row = (long)blockIdx.x * 8 + w;            // (b*T + t)*nheads + h
  int h   = (int)(row % nheads);
  long bt = row / nheads;
  int t   = (int)(bt % T_);
  int b   = (int)(bt / T_);

  const float* src = in + row * 64;
  float x1 = src[lane], x2 = src[lane + 32];
  float c = cosb[t * 32 + lane], s = sinb[t * 32 + lane];
  float y1 =  x1 * c + x2 * s;
  float y2 = -x1 * s + x2 * c;

  float sum = y1 * y1 + y2 * y2;
#pragma unroll
  for (int m = 16; m >= 1; m >>= 1) sum += __shfl_xor(sum, m, 32);
  float scale = rsqrtf(sum * (1.0f / 64.0f) + 1e-6f) * outscale;

  unsigned short* dst = out + (((size_t)b * nheads + h) * T_ + t) * 64;
  dst[lane]      = f2bf(y1 * scale);
  dst[lane + 32] = f2bf(y2 * scale);
}

// ---------------------------------------------------------------------------
// 5) V: [B,T,KVH,64] fp32 -> [B,KVH,64,T] bf16 (transposed so PV B-fragments
//    stream contiguously along the key axis)
// ---------------------------------------------------------------------------
__global__ __launch_bounds__(256) void vtrans_kernel(
    const float* __restrict__ vf, unsigned short* __restrict__ vt) {
  long i = (long)blockIdx.x * 256 + threadIdx.x;   // over B*T*KVH*64
  if (i >= (long)B_ * T_ * KVH * 64) return;
  int d   = (int)(i & 63);
  long r  = i >> 6;
  int h   = (int)(r % KVH);
  long bt = r / KVH;
  int t   = (int)(bt % T_);
  int b   = (int)(bt / T_);
  vt[(((size_t)b * KVH + h) * 64 + d) * T_ + t] = f2bf(vf[i]);
}

// ---------------------------------------------------------------------------
// 6) Flash attention (causal, online softmax, GQA).
//    One wave per 16-query tile of one head. Q pre-scaled by 1/8*log2e.
//    Qb: [B,NH,T,64] bf16   Kb: [B,KVH,T,64] bf16   Vt: [B,KVH,64,T] bf16
//    Yb: [B,T,NH,64] bf16 (== [B*T, C] row-major for the Wo GEMM)
// ---------------------------------------------------------------------------
__global__ __launch_bounds__(32) void flash_kernel(
    const unsigned short* __restrict__ Qb, const unsigned short* __restrict__ Kb,
    const unsigned short* __restrict__ Vt, unsigned short* __restrict__ Yb) {
  __shared__ __align__(16) unsigned short Plds[16 * LDP];

  const int lane = threadIdx.x;
  const int half = lane >> 4;
  const int l16  = lane & 15;

  const int Tt = T_ / 16;
  int bid = blockIdx.x;
  int qt = bid % Tt;
  int h  = (bid / Tt) % NH;
  int b  = bid / (Tt * NH);
  int t0 = qt * 16;
  int kvh = h / (NH / KVH);

  // Q fragments: rows t0..t0+15, d split 0..31 / 32..63
  const unsigned short* qrow = Qb + ((size_t)(b * NH + h) * T_ + (t0 + l16)) * 64;
  v16bf qa0 = mkfrag(ld4(qrow + half * 8),      ld4(qrow + 16 + half * 8));
  v16bf qa1 = mkfrag(ld4(qrow + 32 + half * 8), ld4(qrow + 48 + half * 8));

  const unsigned short* Kbase = Kb + (size_t)(b * KVH + kvh) * T_ * 64;
  const unsigned short* Vbase = Vt + (size_t)(b * KVH + kvh) * 64 * T_;

  v8f acc0 = {}, acc1 = {}, acc2 = {}, acc3 = {};
  float mrow[8], lrow[8];
#pragma unroll
  for (int r = 0; r < 8; r++) { mrow[r] = -__builtin_inff(); lrow[r] = 0.0f; }

  const int tmax = t0 + 15;
  for (int s0 = 0; s0 <= tmax; s0 += 32) {
    // K fragments (B-operand): keys s0..+15 and s0+16..+31, each over d=0..63
    const unsigned short* k0 = Kbase + (size_t)(s0 + l16) * 64 + half * 8;
    v16bf kb0lo = mkfrag(ld4(k0),      ld4(k0 + 16));
    v16bf kb0hi = mkfrag(ld4(k0 + 32), ld4(k0 + 48));
    const unsigned short* k1 = k0 + 16 * 64;
    v16bf kb1lo = mkfrag(ld4(k1),      ld4(k1 + 16));
    v16bf kb1hi = mkfrag(ld4(k1 + 32), ld4(k1 + 48));

    v8f z = {};
    v8f S0 = wmma_bf16(qa0, kb0lo, z); S0 = wmma_bf16(qa1, kb0hi, S0);
    v8f S1 = wmma_bf16(qa0, kb1lo, z); S1 = wmma_bf16(qa1, kb1hi, S1);

    // causal mask + per-row chunk max (logits already in log2 domain)
    float p0[8], p1[8], mc[8];
#pragma unroll
    for (int r = 0; r < 8; r++) {
      int trow = t0 + r + 8 * half;
      float a = ((s0 + l16)      <= trow) ? S0[r] : -1e9f;
      float c = ((s0 + 16 + l16) <= trow) ? S1[r] : -1e9f;
      p0[r] = a; p1[r] = c;
      mc[r] = fmaxf(a, c);
    }
#pragma unroll
    for (int msk = 8; msk >= 1; msk >>= 1)
#pragma unroll
      for (int r = 0; r < 8; r++) mc[r] = fmaxf(mc[r], __shfl_xor(mc[r], msk, 32));

    float alpha[8], rs[8];
#pragma unroll
    for (int r = 0; r < 8; r++) {
      float mn = fmaxf(mrow[r], mc[r]);
      alpha[r] = exp2f(mrow[r] - mn);
      mrow[r] = mn;
      float e0 = exp2f(p0[r] - mn);
      float e1 = exp2f(p1[r] - mn);
      rs[r] = e0 + e1;
      int row = r + 8 * half;
      Plds[row * LDP + l16]      = f2bf(e0);
      Plds[row * LDP + 16 + l16] = f2bf(e1);
    }
#pragma unroll
    for (int msk = 8; msk >= 1; msk >>= 1)
#pragma unroll
      for (int r = 0; r < 8; r++) rs[r] += __shfl_xor(rs[r], msk, 32);
#pragma unroll
    for (int r = 0; r < 8; r++) lrow[r] = lrow[r] * alpha[r] + rs[r];

    __syncthreads();
    // P as A-fragment (16 queries x 32 keys)
    const unsigned short* pr = Plds + l16 * LDP;
    v16bf pa = mkfrag(ld4(pr + half * 8), ld4(pr + 16 + half * 8));
    __syncthreads();

#pragma unroll
    for (int r = 0; r < 8; r++) {
      acc0[r] *= alpha[r]; acc1[r] *= alpha[r];
      acc2[r] *= alpha[r]; acc3[r] *= alpha[r];
    }
    // V fragments: 4 d-column tiles, keys contiguous (Vt layout)
    const unsigned short* v0 = Vbase + (size_t)(l16) * T_ + s0 + half * 8;
    acc0 = wmma_bf16(pa, mkfrag(ld4(v0), ld4(v0 + 16)), acc0);
    const unsigned short* v1 = v0 + (size_t)16 * T_;
    acc1 = wmma_bf16(pa, mkfrag(ld4(v1), ld4(v1 + 16)), acc1);
    const unsigned short* v2 = v0 + (size_t)32 * T_;
    acc2 = wmma_bf16(pa, mkfrag(ld4(v2), ld4(v2 + 16)), acc2);
    const unsigned short* v3 = v0 + (size_t)48 * T_;
    acc3 = wmma_bf16(pa, mkfrag(ld4(v3), ld4(v3 + 16)), acc3);
  }

  // epilogue: normalize and store y[b][t][h][d] bf16
#pragma unroll
  for (int r = 0; r < 8; r++) {
    float inv = 1.0f / lrow[r];
    int t = t0 + r + 8 * half;
    unsigned short* dst = Yb + (((size_t)b * T_ + t) * NH + h) * 64;
    dst[l16]      = f2bf(acc0[r] * inv);
    dst[16 + l16] = f2bf(acc1[r] * inv);
    dst[32 + l16] = f2bf(acc2[r] * inv);
    dst[48 + l16] = f2bf(acc3[r] * inv);
  }
}

// ---------------------------------------------------------------------------
// Launch
// ---------------------------------------------------------------------------
extern "C" void kernel_launch(void* const* d_in, const int* in_sizes, int n_in,
                              void* d_out, int out_size, void* d_ws, size_t ws_size,
                              hipStream_t stream) {
  const float* x    = (const float*)d_in[0];
  const float* cosb = (const float*)d_in[1];
  const float* sinb = (const float*)d_in[2];
  const float* Wq   = (const float*)d_in[3];
  const float* Wk   = (const float*)d_in[4];
  const float* Wv   = (const float*)d_in[5];
  const float* Wo   = (const float*)d_in[6];
  float* out = (float*)d_out;

  char* ws = (char*)d_ws;
  size_t off = 0;
  auto carve = [&](size_t bytes) -> char* {
    char* p = ws + off;
    off = (off + bytes + 255) & ~(size_t)255;
    return p;
  };
  unsigned short* x_bf = (unsigned short*)carve((size_t)MROWS * C_ * 2);        // 8 MiB
  unsigned short* Wq_t = (unsigned short*)carve((size_t)C_ * C_ * 2);           // 2 MiB
  unsigned short* Wk_t = (unsigned short*)carve((size_t)C_ * (KVH * HD) * 2);
  unsigned short* Wv_t = (unsigned short*)carve((size_t)C_ * (KVH * HD) * 2);
  unsigned short* Wo_t = (unsigned short*)carve((size_t)C_ * C_ * 2);
  float* qf = (float*)carve((size_t)MROWS * C_ * 4);                            // 16 MiB
  float* kf = (float*)carve((size_t)MROWS * (KVH * HD) * 4);
  float* vf = (float*)carve((size_t)MROWS * (KVH * HD) * 4);
  unsigned short* Qb = (unsigned short*)carve((size_t)MROWS * C_ * 2);
  unsigned short* Kb = (unsigned short*)carve((size_t)MROWS * (KVH * HD) * 2);
  unsigned short* Vt = (unsigned short*)carve((size_t)MROWS * (KVH * HD) * 2);
  unsigned short* y_bf = (unsigned short*)carve((size_t)MROWS * C_ * 2);
  (void)ws_size; (void)in_sizes; (void)n_in; (void)out_size;

  // 1) casts / weight transposes
  {
    long n = (long)MROWS * C_;
    cast_bf16_kernel<<<(unsigned)((n + 255) / 256), 256, 0, stream>>>(x, x_bf, n);
  }
  wtrans_kernel<<<(unsigned)(((long)C_ * C_ + 255) / 256), 256, 0, stream>>>(Wq, Wq_t, C_, C_);
  wtrans_kernel<<<(unsigned)(((long)C_ * KVH * HD + 255) / 256), 256, 0, stream>>>(Wk, Wk_t, C_, KVH * HD);
  wtrans_kernel<<<(unsigned)(((long)C_ * KVH * HD + 255) / 256), 256, 0, stream>>>(Wv, Wv_t, C_, KVH * HD);
  wtrans_kernel<<<(unsigned)(((long)C_ * C_ + 255) / 256), 256, 0, stream>>>(Wo, Wo_t, C_, C_);

  // 2) QKV projections (WMMA GEMMs, TDM-staged tiles)
  gemm_bf16_tn_kernel<<<dim3(C_ / GBN, MROWS / GBM), 256, 0, stream>>>(x_bf, Wq_t, qf, MROWS, C_, C_);
  gemm_bf16_tn_kernel<<<dim3((KVH * HD) / GBN, MROWS / GBM), 256, 0, stream>>>(x_bf, Wk_t, kf, MROWS, KVH * HD, C_);
  gemm_bf16_tn_kernel<<<dim3((KVH * HD) / GBN, MROWS / GBM), 256, 0, stream>>>(x_bf, Wv_t, vf, MROWS, KVH * HD, C_);

  // 3) RoPE + RMSNorm + head transpose; fold 1/sqrt(64)*log2(e) into Q
  const float QSCALE = 0.125f * 1.4426950408889634f;
  rope_rms_kernel<<<(unsigned)((long)MROWS * NH / 8), 256, 0, stream>>>(qf, cosb, sinb, Qb, NH, QSCALE);
  rope_rms_kernel<<<(unsigned)((long)MROWS * KVH / 8), 256, 0, stream>>>(kf, cosb, sinb, Kb, KVH, 1.0f);
  vtrans_kernel<<<(unsigned)(((long)MROWS * KVH * HD + 255) / 256), 256, 0, stream>>>(vf, Vt);

  // 4) flash attention: one wave per 16-query tile per head
  flash_kernel<<<(unsigned)(B_ * NH * (T_ / 16)), 32, 0, stream>>>(Qb, Kb, Vt, y_bf);

  // 5) output projection -> fp32 out
  gemm_bf16_tn_kernel<<<dim3(C_ / GBN, MROWS / GBM), 256, 0, stream>>>(y_bf, Wo_t, out, MROWS, C_, C_);
}
